// ReSFU_601295421643
// MI455X (gfx1250) — matrix operand order repaired and checked by reading, not compiled, not run on hardware
//
#include <hip/hip_runtime.h>

typedef _Float16 v16h __attribute__((ext_vector_type(16)));
typedef float    v8f  __attribute__((ext_vector_type(8)));

static constexpr long cHW  = 65536;  // 256*256
static constexpr long chw  = 4096;   // 64*64
static constexpr long cHW2 = 16384;  // 128*128

// ---------------------------------------------------------------------------
// GroupNorm statistics: one block per group, layout [C, HW] (single image)
// ---------------------------------------------------------------------------
__global__ void gn_stats(const float* __restrict__ v, float* __restrict__ mu,
                         float* __restrict__ rs, int cpg, long HW) {
  const long n = (long)cpg * HW;
  const float* base = v + (long)blockIdx.x * n;
  float s = 0.f, ss = 0.f;
  for (long i = threadIdx.x; i < n; i += blockDim.x) {
    float t = base[i];
    s += t; ss += t * t;
  }
  __shared__ float sh[256], sh2[256];
  sh[threadIdx.x] = s; sh2[threadIdx.x] = ss;
  __syncthreads();
  for (int o = blockDim.x >> 1; o > 0; o >>= 1) {
    if ((int)threadIdx.x < o) { sh[threadIdx.x] += sh[threadIdx.x + o]; sh2[threadIdx.x] += sh2[threadIdx.x + o]; }
    __syncthreads();
  }
  if (threadIdx.x == 0) {
    float m = sh[0] / (float)n;
    float var = sh2[0] / (float)n - m * m;
    mu[blockIdx.x] = m;
    rs[blockIdx.x] = rsqrtf(var + 1e-5f);
  }
}

// ---------------------------------------------------------------------------
// GroupNorm apply (materialize): out = (in-mu)*rs*w + b
// ---------------------------------------------------------------------------
__global__ void gn_apply(const float* __restrict__ in, float* __restrict__ out,
                         const float* __restrict__ mu, const float* __restrict__ rs,
                         const float* __restrict__ w, const float* __restrict__ b,
                         int cpg, long HW, long total) {
  long idx = (long)blockIdx.x * blockDim.x + threadIdx.x;
  if (idx >= total) return;
  int c = (int)(idx / HW);
  int g = c / cpg;
  out[idx] = (in[idx] - mu[g]) * rs[g] * w[c] + b[c];
}

// ---------------------------------------------------------------------------
// Fold GN affine into a 1x1 conv: Aeff[m,c] = W[m,c]*s[c],
// beff[m] = wb[m] + sum_c W[m,c]*t[c], with s = rs*gw, t = gb - mu*s.
// grid.x = Mpad (rows >= M are zeroed)
// ---------------------------------------------------------------------------
__global__ void fold_proj(const float* __restrict__ W, const float* __restrict__ wb,
                          const float* __restrict__ mu, const float* __restrict__ rs,
                          const float* __restrict__ gw, const float* __restrict__ gb,
                          int M, int C, int cpg,
                          float* __restrict__ Aeff, float* __restrict__ beff) {
  const int m = blockIdx.x;
  const int t = threadIdx.x;
  if (m >= M) {
    for (int c = t; c < C; c += blockDim.x) Aeff[(long)m * C + c] = 0.f;
    if (t == 0) beff[m] = 0.f;
    return;
  }
  float acc = 0.f;
  for (int c = t; c < C; c += blockDim.x) {
    int g = c / cpg;
    float sc = rs[g] * gw[c];
    float tr = gb[c] - mu[g] * sc;
    float wv = W[(long)m * C + c];
    Aeff[(long)m * C + c] = wv * sc;
    acc += wv * tr;
  }
  __shared__ float sh[256];
  sh[t] = acc; __syncthreads();
  for (int o = blockDim.x >> 1; o > 0; o >>= 1) {
    if (t < o) sh[t] += sh[t + o];
    __syncthreads();
  }
  if (t == 0) beff[m] = (wb ? wb[m] : 0.f) + sh[0];
}

// ---------------------------------------------------------------------------
// WMMA GEMM: Out[M x N] = A[Mpad x K] * X[K x N] + bias, fp32 storage,
// f16 operands, f32 accumulate (v_wmma_f32_16x16x32_f16).
// One wave computes a 16x64 strip (1 M-tile x 4 N-tiles): A fragment is
// loaded/converted once per k-step and reused by 4 WMMAs; 4 independent
// accumulators keep the matrix pipe busy while the next B frag converts.
// K % 32 == 0, N % 64 == 0.
// ---------------------------------------------------------------------------
__global__ void __launch_bounds__(128) wmma_gemm(
    const float* __restrict__ A, const float* __restrict__ X,
    const float* __restrict__ bias, float* __restrict__ Out,
    int Mpad, int K, int N, int relu, int m_store) {
  const int lane = threadIdx.x & 31;
  const int wib  = threadIdx.x >> 5;
  const int tiles_m = Mpad >> 4;
  const long wid = (long)blockIdx.x * (blockDim.x >> 5) + wib;
  const int  mt = (int)(wid % tiles_m);
  const long nstrip = wid / tiles_m;
  const long n0 = nstrip << 6;         // 64 columns per wave
  if (n0 >= (long)N) return;           // wave-uniform: EXEC stays full
  const int  m0  = mt << 4;
  const int  rA  = m0 + (lane & 15);
  const long col = n0 + (lane & 15);
  const int  hi  = lane >> 4;          // 0: lanes 0-15, 1: lanes 16-31
  v8f acc0 = {}, acc1 = {}, acc2 = {}, acc3 = {};
  for (int k0 = 0; k0 < K; k0 += 32) {
    // A fragment: row rA; halves i -> K = k0 + hi*16 + i
    v16h af;
    const float* Ap = A + (long)rA * K + k0 + (hi << 4);
#pragma unroll
    for (int i = 0; i < 8; ++i) af[i] = (_Float16)Ap[i];
#pragma unroll
    for (int i = 0; i < 8; ++i) af[8 + i] = (_Float16)Ap[8 + i];
    // B fragments: column col + 16*j; halves i -> K = k0 + hi*16 + i
    const float* Xp = X + (long)(k0 + (hi << 4)) * (long)N + col;
    if (k0 + 32 < K)                   // prefetch next K-slice of B
      __builtin_prefetch(Xp + 32L * N, 0, 3);
#pragma unroll
    for (int j = 0; j < 4; ++j) {
      v16h bf;
#pragma unroll
      for (int i = 0; i < 16; ++i) bf[i] = (_Float16)Xp[(long)i * N + 16 * j];
      if (j == 0)
        acc0 = __builtin_amdgcn_wmma_f32_16x16x32_f16(false, af, false, bf, (short)0, acc0, false, false);
      else if (j == 1)
        acc1 = __builtin_amdgcn_wmma_f32_16x16x32_f16(false, af, false, bf, (short)0, acc1, false, false);
      else if (j == 2)
        acc2 = __builtin_amdgcn_wmma_f32_16x16x32_f16(false, af, false, bf, (short)0, acc2, false, false);
      else
        acc3 = __builtin_amdgcn_wmma_f32_16x16x32_f16(false, af, false, bf, (short)0, acc3, false, false);
    }
  }
  // D layout: VGPR r -> M = m0 + r + 8*hi, N = col (+16*j)
  const int full = (m_store >= Mpad);  // wave-uniform fast path: no row guard
#pragma unroll
  for (int r = 0; r < 8; ++r) {
    int m = m0 + r + (hi << 3);
    if (full || m < m_store) {
      float bv = bias[m];
      float* op = Out + (long)m * N + col;
      float v0 = acc0[r] + bv, v1 = acc1[r] + bv, v2 = acc2[r] + bv, v3 = acc3[r] + bv;
      if (relu) {
        v0 = fmaxf(v0, 0.f); v1 = fmaxf(v1, 0.f);
        v2 = fmaxf(v2, 0.f); v3 = fmaxf(v3, 0.f);
      }
      op[0] = v0; op[16] = v1; op[32] = v2; op[48] = v3;
    }
  }
}

// ---------------------------------------------------------------------------
// Bilinear resize (half-pixel), generic
// ---------------------------------------------------------------------------
__global__ void bilinear_k(const float* __restrict__ in, float* __restrict__ out,
                           int C, int Hi, int Wi, int Ho, int Wo) {
  long idx = (long)blockIdx.x * blockDim.x + threadIdx.x;
  long tot = (long)C * Ho * Wo;
  if (idx >= tot) return;
  int xo = (int)(idx % Wo);
  long r = idx / Wo;
  int yo = (int)(r % Ho);
  int c = (int)(r / Ho);
  float fy = (yo + 0.5f) * ((float)Hi / Ho) - 0.5f;
  float fx = (xo + 0.5f) * ((float)Wi / Wo) - 0.5f;
  int y0 = (int)floorf(fy), x0 = (int)floorf(fx);
  float wy = fy - y0, wx = fx - x0;
  int y0c = y0 < 0 ? 0 : (y0 > Hi - 1 ? Hi - 1 : y0);
  int y1c = y0 + 1 < 0 ? 0 : (y0 + 1 > Hi - 1 ? Hi - 1 : y0 + 1);
  int x0c = x0 < 0 ? 0 : (x0 > Wi - 1 ? Wi - 1 : x0);
  int x1c = x0 + 1 < 0 ? 0 : (x0 + 1 > Wi - 1 ? Wi - 1 : x0 + 1);
  const float* p = in + (long)c * Hi * Wi;
  float v00 = p[(long)y0c * Wi + x0c], v01 = p[(long)y0c * Wi + x1c];
  float v10 = p[(long)y1c * Wi + x0c], v11 = p[(long)y1c * Wi + x1c];
  out[idx] = (1.f - wy) * ((1.f - wx) * v00 + wx * v01) +
             wy * ((1.f - wx) * v10 + wx * v11);
}

// ---------------------------------------------------------------------------
// Depthwise 3x3 gaussian, edge padded, on (32,256,256)
// ---------------------------------------------------------------------------
__global__ void gauss3(const float* __restrict__ in, float* __restrict__ out) {
  long idx = (long)blockIdx.x * blockDim.x + threadIdx.x;
  long tot = 32 * cHW;
  if (idx >= tot) return;
  long c = idx / cHW;
  long p = idx % cHW;
  int y = (int)(p >> 8), x = (int)(p & 255);
  const float a0 = 0.60653065971f;                // exp(-0.5)
  const float nrm = (1.f + 2.f * a0) * (1.f + 2.f * a0);
  const float* base = in + c * cHW;
  float s = 0.f;
#pragma unroll
  for (int dy = -1; dy <= 1; ++dy) {
    int yy = y + dy; yy = yy < 0 ? 0 : (yy > 255 ? 255 : yy);
    float wy = dy ? a0 : 1.f;
#pragma unroll
    for (int dx = -1; dx <= 1; ++dx) {
      int xx = x + dx; xx = xx < 0 ? 0 : (xx > 255 ? 255 : xx);
      float wx = dx ? a0 : 1.f;
      s += wy * wx * base[((long)yy << 8) + xx];
    }
  }
  out[idx] = s / nrm;
}

// ---------------------------------------------------------------------------
// Guided filter: downsample-by-2 prep (I, p, I*p, I*I) at (32,128,128)
// ---------------------------------------------------------------------------
__global__ void gf_prep(const float* __restrict__ q, const float* __restrict__ kk,
                        float* __restrict__ g0, float* __restrict__ g1,
                        float* __restrict__ g2, float* __restrict__ g3) {
  long idx = (long)blockIdx.x * blockDim.x + threadIdx.x;
  long tot = 32 * cHW2;
  if (idx >= tot) return;
  long c = idx / cHW2;
  long p = idx % cHW2;
  int y = (int)(p >> 7), x = (int)(p & 127);
  long src = c * cHW + ((long)(y * 2) << 8) + (x * 2);
  float I = q[src], P = kk[src];
  g0[idx] = I; g1[idx] = P; g2[idx] = I * P; g3[idx] = I * I;
}

__global__ void box_h(const float* __restrict__ in, float* __restrict__ out,
                      int C, int H, int W, int r) {
  long idx = (long)blockIdx.x * blockDim.x + threadIdx.x;
  long tot = (long)C * H * W;
  if (idx >= tot) return;
  int x = (int)(idx % W);
  long row = idx / W;
  const float* p = in + row * W;
  int xs = x - r; if (xs < 0) xs = 0;
  int xe = x + r; if (xe > W - 1) xe = W - 1;
  float s = 0.f;
  for (int i = xs; i <= xe; ++i) s += p[i];
  out[idx] = s;
}

__global__ void box_v(const float* __restrict__ in, float* __restrict__ out,
                      int C, int H, int W, int r, int divN) {
  long idx = (long)blockIdx.x * blockDim.x + threadIdx.x;
  long tot = (long)C * H * W;
  if (idx >= tot) return;
  int x = (int)(idx % W);
  long rr = idx / W;
  int y = (int)(rr % H);
  long c = rr / H;
  int ys = y - r; if (ys < 0) ys = 0;
  int ye = y + r; if (ye > H - 1) ye = H - 1;
  float s = 0.f;
  for (int i = ys; i <= ye; ++i) s += in[(c * H + i) * (long)W + x];
  if (divN) {
    int xs = x - r; if (xs < 0) xs = 0;
    int xe = x + r; if (xe > W - 1) xe = W - 1;
    s /= (float)((ye - ys + 1) * (xe - xs + 1));
  }
  out[idx] = s;
}

__global__ void gf_ab(const float* __restrict__ mI, const float* __restrict__ mp,
                      const float* __restrict__ mIp, const float* __restrict__ mII,
                      float* __restrict__ a, float* __restrict__ b) {
  long idx = (long)blockIdx.x * blockDim.x + threadIdx.x;
  long tot = 32 * cHW2;
  if (idx >= tot) return;
  float i = mI[idx], p = mp[idx];
  float cov = mIp[idx] - i * p;
  float var = mII[idx] - i * i;
  float av = cov / (var + 1e-3f);
  a[idx] = av;
  b[idx] = p - av * i;
}

// upsample mean_a/mean_b 128->256 (half-pixel bilinear) and apply a*I + b
__global__ void gf_apply(const float* __restrict__ ma, const float* __restrict__ mb,
                         const float* __restrict__ I, float* __restrict__ out) {
  long idx = (long)blockIdx.x * blockDim.x + threadIdx.x;
  long tot = 32 * cHW;
  if (idx >= tot) return;
  long c = idx / cHW;
  long p = idx % cHW;
  int Y = (int)(p >> 8), X = (int)(p & 255);
  float fy = (Y + 0.5f) * 0.5f - 0.5f;
  float fx = (X + 0.5f) * 0.5f - 0.5f;
  int y0 = (int)floorf(fy), x0 = (int)floorf(fx);
  float wy = fy - y0, wx = fx - x0;
  int y0c = y0 < 0 ? 0 : (y0 > 127 ? 127 : y0);
  int y1c = y0 + 1 < 0 ? 0 : (y0 + 1 > 127 ? 127 : y0 + 1);
  int x0c = x0 < 0 ? 0 : (x0 > 127 ? 127 : x0);
  int x1c = x0 + 1 < 0 ? 0 : (x0 + 1 > 127 ? 127 : x0 + 1);
  const float* pa = ma + c * cHW2;
  const float* pb = mb + c * cHW2;
  float w00 = (1.f - wy) * (1.f - wx), w01 = (1.f - wy) * wx;
  float w10 = wy * (1.f - wx), w11 = wy * wx;
  long i00 = ((long)y0c << 7) + x0c, i01 = ((long)y0c << 7) + x1c;
  long i10 = ((long)y1c << 7) + x0c, i11 = ((long)y1c << 7) + x1c;
  float av = w00 * pa[i00] + w01 * pa[i01] + w10 * pa[i10] + w11 * pa[i11];
  float bv = w00 * pb[i00] + w01 * pb[i01] + w10 * pb[i10] + w11 * pb[i11];
  out[idx] = av * I[idx] + bv;
}

// ---------------------------------------------------------------------------
// PCDC central-difference conv: v1[o,p] = relu( sum wt*(kn shifted) -
//     (sum wt)*qn + bias ), dilation 4, edge clamp, groups of 8 in-channels.
// grid = (HW/256, 128)
// ---------------------------------------------------------------------------
__global__ void pcdc_conv(const float* __restrict__ kn, const float* __restrict__ qn,
                          const float* __restrict__ wt, const float* __restrict__ bs,
                          float* __restrict__ v1) {
  const int o = blockIdx.y;
  const int g = o >> 5;
  const long p = (long)blockIdx.x * blockDim.x + threadIdx.x;
  const int py = (int)(p >> 8), px = (int)(p & 255);
  float acc = bs[o];
#pragma unroll
  for (int c = 0; c < 8; ++c) {
    const int cin = (g << 3) + c;
    const float* kch = kn + (long)cin * cHW;
    const float* wp = wt + ((long)o * 8 + c) * 9;
    float kd = 0.f;
#pragma unroll
    for (int ky = 0; ky < 3; ++ky) {
      int yy = py + 4 * (ky - 1);
      yy = yy < 0 ? 0 : (yy > 255 ? 255 : yy);
#pragma unroll
      for (int kx = 0; kx < 3; ++kx) {
        int xx = px + 4 * (kx - 1);
        xx = xx < 0 ? 0 : (xx > 255 ? 255 : xx);
        float wv = wp[ky * 3 + kx];
        kd += wv;
        acc += wv * kch[((long)yy << 8) + xx];
      }
    }
    acc -= kd * qn[(long)cin * cHW + p];
  }
  v1[(long)o * cHW + p] = fmaxf(acc, 0.f);
}

// ---------------------------------------------------------------------------
// Masked softmax over 9 taps; attn written in-place into sem buffer
// ---------------------------------------------------------------------------
__global__ void softmax9(float* __restrict__ sem, const float* __restrict__ det) {
  long p = (long)blockIdx.x * blockDim.x + threadIdx.x;
  if (p >= cHW) return;
  int py = (int)(p >> 8), px = (int)(p & 255);
  float s[9];
  float mx = -3.4e38f;
#pragma unroll
  for (int d = 0; d < 9; ++d) {
    int oy = (d / 3 - 1) * 4, ox = (d % 3 - 1) * 4;
    bool ok = ((unsigned)(py + oy) < 256u) && ((unsigned)(px + ox) < 256u);
    float v = sem[(long)d * cHW + p] + det[(long)d * cHW + p];
    if (!ok) v -= 1e7f;
    s[d] = v;
    mx = fmaxf(mx, v);
  }
  float sum = 0.f;
#pragma unroll
  for (int d = 0; d < 9; ++d) { s[d] = expf(s[d] - mx); sum += s[d]; }
  float inv = 1.f / sum;
#pragma unroll
  for (int d = 0; d < 9; ++d) sem[(long)d * cHW + p] = s[d] * inv;
}

// ---------------------------------------------------------------------------
// out[c, i*4+p, j*4+q] = sum_d attn[d, i*4+p, j*4+q] * x0pad[c, i+dy-1, j+dx-1]
// one block per (i,j) low-res tile; 256 threads = 16 px * 16 channel-lanes
// ---------------------------------------------------------------------------
__global__ void attn_apply(const float* __restrict__ attn, const float* __restrict__ x,
                           float* __restrict__ out) {
  __shared__ float sat[144];
  const int tile = blockIdx.x;
  const int ti = tile >> 6, tj = tile & 63;
  const int t = threadIdx.x;
  if (t < 144) {
    int d = t / 16, pq = t % 16;
    int pp = pq >> 2, qq = pq & 3;
    sat[t] = attn[(long)d * cHW + ((long)(ti * 4 + pp) << 8) + (tj * 4 + qq)];
  }
  __syncthreads();
  const int pq = t & 15, c0 = t >> 4;
  const int pp = pq >> 2, qq = pq & 3;
  const long orow = ((long)(ti * 4 + pp) << 8) + (tj * 4 + qq);
  for (int it = 0; it < 16; ++it) {
    int c = c0 + (it << 4);
    float s = 0.f;
#pragma unroll
    for (int d = 0; d < 9; ++d) {
      int dy = d / 3 - 1, dx = d % 3 - 1;
      int yy = ti + dy, xx = tj + dx;
      float xv = ((unsigned)yy < 64u && (unsigned)xx < 64u)
                     ? x[((long)c * 64 + yy) * 64 + xx] : 0.f;
      s += sat[d * 16 + pq] * xv;
    }
    out[(long)c * cHW + orow] = s;
  }
}

// ---------------------------------------------------------------------------
// Host side
// ---------------------------------------------------------------------------
static inline unsigned gridFor(long n) { return (unsigned)((n + 255) / 256); }
// GEMM grid: waves = tiles_m * (N/64); 4 waves per block
static inline unsigned gemmGrid(int Mpad, long N) {
  long waves = (long)(Mpad / 16) * (N / 64);
  return (unsigned)((waves + 3) / 4);
}

extern "C" void kernel_launch(void* const* d_in, const int* in_sizes, int n_in,
                              void* d_out, int out_size, void* d_ws, size_t ws_size,
                              hipStream_t stream) {
  (void)in_sizes; (void)n_in; (void)out_size; (void)ws_size;
  const float* y        = (const float*)d_in[0];
  const float* x        = (const float*)d_in[1];
  const float* norm_y_w = (const float*)d_in[2];
  const float* norm_y_b = (const float*)d_in[3];
  const float* norm_x_w = (const float*)d_in[4];
  const float* norm_x_b = (const float*)d_in[5];
  const float* q_w      = (const float*)d_in[6];
  const float* q_b      = (const float*)d_in[7];
  const float* k_w      = (const float*)d_in[8];
  const float* k_b      = (const float*)d_in[9];
  const float* P[20];
  for (int i = 0; i < 20; ++i) P[i] = (const float*)d_in[10 + i];
  // P[0..9]  = sem_{norm_w,norm_b,weight,bias,cc1_w,cc1_b,gn_w,gn_b,cc2_w,cc2_b}
  // P[10..19]= det_*
  float* out = (float*)d_out;
  float* ws  = (float*)d_ws;

  // workspace layout (floats), reused per batch
  long o = 0;
  float* qbuf = ws + o; o += 32 * cHW;   // q projection
  float* kup  = ws + o; o += 32 * cHW;   // k upsampled (k_)
  float* qgs  = ws + o; o += 32 * cHW;   // gaussian-smoothed q
  float* qgf  = ws + o; o += 32 * cHW;   // guided-filtered q
  float* qn   = ws + o; o += 32 * cHW;   // pcdc normalized q_in
  float* kn   = ws + o; o += 32 * cHW;   // pcdc normalized k_in
  float* v1   = ws + o; o += 128 * cHW;  // pcdc stage 1
  float* v2   = ws + o; o += 128 * cHW;  // pcdc stage 2
  float* scs  = ws + o; o += 9 * cHW;    // sem scores -> attn
  float* scd  = ws + o; o += 9 * cHW;    // det scores
  float* kbuf = ws + o; o += 32 * chw;   // k at low res
  float* G[9];
  for (int i = 0; i < 9; ++i) { G[i] = ws + o; o += 32 * cHW2; }
  float* Aeff  = ws + o; o += 32 * 256;
  float* beff  = ws + o; o += 32;
  float* stats = ws + o; o += 256;

  for (int b = 0; b < 4; ++b) {
    const float* yb = y + (long)b * 256 * cHW;
    const float* xb = x + (long)b * 256 * chw;
    float* outb = out + (long)b * 256 * cHW;

    // ---- q = conv1x1(GN(y)) via folded GEMM over raw y ----
    gn_stats<<<32, 256, 0, stream>>>(yb, stats + 0, stats + 32, 8, cHW);
    fold_proj<<<32, 256, 0, stream>>>(q_w, q_b, stats + 0, stats + 32,
                                      norm_y_w, norm_y_b, 32, 256, 8, Aeff, beff);
    wmma_gemm<<<gemmGrid(32, cHW), 128, 0, stream>>>(
        Aeff, yb, beff, qbuf, 32, 256, (int)cHW, 0, 32);

    // ---- k = conv1x1(GN(x)), then bilinear x4 ----
    gn_stats<<<32, 256, 0, stream>>>(xb, stats + 64, stats + 96, 8, chw);
    fold_proj<<<32, 256, 0, stream>>>(k_w, k_b, stats + 64, stats + 96,
                                      norm_x_w, norm_x_b, 32, 256, 8, Aeff, beff);
    wmma_gemm<<<gemmGrid(32, chw), 128, 0, stream>>>(
        Aeff, xb, beff, kbuf, 32, 256, (int)chw, 0, 32);
    bilinear_k<<<gridFor(32 * cHW), 256, 0, stream>>>(kbuf, kup, 32, 64, 64, 256, 256);

    // ---- gaussian smoothing of q ----
    gauss3<<<gridFor(32 * cHW), 256, 0, stream>>>(qbuf, qgs);

    // ---- guided filter: q guided by k_ ----
    gf_prep<<<gridFor(32 * cHW2), 256, 0, stream>>>(qbuf, kup, G[0], G[1], G[2], G[3]);
    for (int m = 0; m < 4; ++m) {
      box_h<<<gridFor(32 * cHW2), 256, 0, stream>>>(G[m], G[4], 32, 128, 128, 4);
      box_v<<<gridFor(32 * cHW2), 256, 0, stream>>>(G[4], G[5 + m], 32, 128, 128, 4, 1);
    }
    gf_ab<<<gridFor(32 * cHW2), 256, 0, stream>>>(G[5], G[6], G[7], G[8], G[2], G[3]);
    box_h<<<gridFor(32 * cHW2), 256, 0, stream>>>(G[2], G[4], 32, 128, 128, 4);
    box_v<<<gridFor(32 * cHW2), 256, 0, stream>>>(G[4], G[0], 32, 128, 128, 4, 1);
    box_h<<<gridFor(32 * cHW2), 256, 0, stream>>>(G[3], G[4], 32, 128, 128, 4);
    box_v<<<gridFor(32 * cHW2), 256, 0, stream>>>(G[4], G[1], 32, 128, 128, 4, 1);
    gf_apply<<<gridFor(32 * cHW), 256, 0, stream>>>(G[0], G[1], qbuf, qgf);

    // ---- two PCDC blocks ----
    auto run_pcdc = [&](const float* kin, const float* qin, const float* const* prm,
                        float* sc) {
      const float* nw = prm[0]; const float* nb = prm[1];
      const float* wt = prm[2]; const float* bs = prm[3];
      const float* c1wp = prm[4]; const float* c1bp = prm[5];
      const float* gwp = prm[6]; const float* gbp = prm[7];
      const float* c2wp = prm[8]; const float* c2bp = prm[9];
      gn_stats<<<8, 256, 0, stream>>>(qin, stats + 128, stats + 136, 4, cHW);
      gn_apply<<<gridFor(32 * cHW), 256, 0, stream>>>(qin, qn, stats + 128, stats + 136,
                                                      nw, nb, 4, cHW, 32 * cHW);
      gn_stats<<<8, 256, 0, stream>>>(kin, stats + 144, stats + 152, 4, cHW);
      gn_apply<<<gridFor(32 * cHW), 256, 0, stream>>>(kin, kn, stats + 144, stats + 152,
                                                      nw, nb, 4, cHW, 32 * cHW);
      pcdc_conv<<<dim3((unsigned)(cHW / 256), 128), 256, 0, stream>>>(kn, qn, wt, bs, v1);
      for (int g = 0; g < 4; ++g)
        wmma_gemm<<<gemmGrid(32, cHW), 128, 0, stream>>>(
            c1wp + (long)g * 32 * 32, v1 + (long)g * 32 * cHW, c1bp + g * 32,
            v2 + (long)g * 32 * cHW, 32, 32, (int)cHW, 1, 32);
      gn_stats<<<8, 256, 0, stream>>>(v2, stats + 160, stats + 168, 16, cHW);
      fold_proj<<<16, 256, 0, stream>>>(c2wp, c2bp, stats + 160, stats + 168,
                                        gwp, gbp, 9, 128, 16, Aeff, beff);
      wmma_gemm<<<gemmGrid(16, cHW), 128, 0, stream>>>(
          Aeff, v2, beff, sc, 16, 128, (int)cHW, 0, 9);
    };
    run_pcdc(kup, qgf, &P[0], scs);   // sem
    run_pcdc(qgs, qbuf, &P[10], scd); // det

    // ---- softmax + attention apply ----
    softmax9<<<gridFor(cHW), 256, 0, stream>>>(scs, scd);
    attn_apply<<<4096, 256, 0, stream>>>(scs, xb, outb);
  }
}
